// HierarchicalNeRF_51745765982747
// MI455X (gfx1250) — matrix operand order repaired and checked by reading, not compiled, not run on hardware
//
#include <hip/hip_runtime.h>
#include <hip/hip_bf16.h>

typedef _Float16 half_t;
typedef __attribute__((ext_vector_type(16))) _Float16 v16h;
typedef __attribute__((ext_vector_type(8)))  _Float16 v8h;
typedef __attribute__((ext_vector_type(8)))  float    v8f;
typedef int v4i __attribute__((ext_vector_type(4)));

#define NUM_RAYS 4096
#define NC 64
#define NFI 128
#define PDIM 123
#define VDIM 51
#define ROWH 392          // padded activation row stride in halfs (384 + 8)

// ---- GEMM table (11 GEMMs per MLP) ------------------------------------
// 0..7 pos layers, 8 density, 9 view0, 10 view1
static constexpr int KPg[11]   = {128,256,256,256,384,256,256,256,256,320,128}; // padded K
static constexpr int NPg[11]   = {256,256,256,256,256,256,256,256, 16,128, 16}; // padded N
static constexpr int DINg[11]  = {123,256,256,256,379,256,256,256,256,307,128};
static constexpr int DOUTg[11] = {256,256,256,256,256,256,256,256,  1,128,  3};
static constexpr int WOFFg[12] = {0,32768,98304,163840,229376,327680,393216,458752,
                                  524288,528384,569344,571392};
static constexpr int BOFFg[12] = {0,256,512,768,1024,1280,1536,1792,2048,2064,2192,2208};
#define WTOTAL 571392   // halfs per MLP
#define BTOTAL 2208     // floats per MLP

// ---- workspace layout (bytes) -----------------------------------------
#define OFF_WPACK 0u
#define OFF_BIAS  2285568u                 // 2*WTOTAL*2
#define OFF_ZC    2303232u
#define OFF_ZF    2303488u
#define OFF_DC    4400640u
#define OFF_RGBC  5449216u
#define OFF_DF    8594944u
#define OFF_RGBF  10692096u
#define WS_TOTAL  16983552u

// ---- LDS layout (halfs): 4 waves per block ----------------------------
#define WBUF_H   65536                 // 128KB shared weight staging buffer
#define ACT_H    6272                  // 16*ROWH per wave
#define LDS_HALFS (WBUF_H + 8*ACT_H + 4*2048 + 4*1024)   // = 128000
#define LDS_BYTES (LDS_HALFS*2)                           // = 256000

static __device__ inline v8f wmma_f16(v16h a, v16h b, v8f c) {
  return __builtin_amdgcn_wmma_f32_16x16x32_f16(false, a, false, b, (short)0, c,
                                                false, false);
}

static __device__ __host__ inline unsigned hash_u(unsigned a, unsigned b) {
  unsigned x = a * 747796405u + b * 2891336453u + 1u;
  x ^= x >> 17; x *= 0xed5ad4bbu; x ^= x >> 11; x *= 0xac4c1b51u; x ^= x >> 15;
  return x;
}
static __device__ inline float rnd01(unsigned a, unsigned b) {
  return (float)(hash_u(a, b) >> 8) * (1.0f / 16777216.0f);
}

// ---- cooperative global->LDS weight staging ---------------------------
// Prefer the CDNA5 async global->LDS path (ASYNCcnt-tracked, no VGPR
// round-trip); fall back to a plain VGPR-bounce copy.
static __device__ __forceinline__ void stage_weights(half_t* __restrict__ wbuf,
                                                     const half_t* __restrict__ wsrc,
                                                     int halfs) {
  const int n16 = halfs >> 3;   // 16-byte chunks
#if __has_builtin(__builtin_amdgcn_global_load_async_to_lds_b128)
  typedef __attribute__((address_space(1))) v4i GV4;   // int4 in global
  typedef __attribute__((address_space(3))) v4i LV4;   // int4 in LDS
  for (int i = threadIdx.x; i < n16; i += blockDim.x) {
    __builtin_amdgcn_global_load_async_to_lds_b128(
        (GV4*)(const_cast<half_t*>(wsrc) + (size_t)i * 8),
        (LV4*)(wbuf + (size_t)i * 8), 0, 0);
  }
#if __has_builtin(__builtin_amdgcn_s_wait_asynccnt)
  __builtin_amdgcn_s_wait_asynccnt(0);
#else
  asm volatile("s_wait_asynccnt 0x0" ::: "memory");
#endif
#else
  const uint4* s = (const uint4*)wsrc;
  uint4* d = (uint4*)wbuf;
  if (threadIdx.x == 0) __builtin_prefetch(s, 0, 1);
  for (int i = threadIdx.x; i < n16; i += blockDim.x) d[i] = s[i];
#endif
}

// =======================================================================
// Per-tile GEMM: one wave computes a 16-row x (NT*16)-col output using
// v_wmma_f32_16x16x32_f16, K streamed in chunks through shared LDS wbuf.
// MODE 0: +bias, ReLU, store f16 to actOut (LDS)
// MODE 1: density head (col 0) -> gout[point]
// MODE 2: rgb head (cols 0..2), sigmoid -> gout[point*3+c]
// =======================================================================
template <int KS, int NT, int MODE>
static __device__ __forceinline__ void gemm_tile(
    const half_t* __restrict__ wsrc, const float* __restrict__ bias,
    const half_t* actIn, half_t* actOut, half_t* wbuf,
    float* gout, int pointBase, int lane) {
  constexpr int CH = (128 / NT < KS) ? (128 / NT) : KS;   // ksteps per LDS chunk
  const int row = lane & 15, hi = lane >> 4;
  v8f acc[NT] = {};

  for (int kc = 0; kc < KS; kc += CH) {
    const int nks = (KS - kc) < CH ? (KS - kc) : CH;
    __syncthreads();   // everyone done with previous wbuf contents
    stage_weights(wbuf, wsrc + (size_t)kc * NT * 512, nks * NT * 512);
    __syncthreads();

#pragma unroll
    for (int ks = 0; ks < CH; ++ks) {
      if (ks < nks) {
        // A fragment (16x32 f16): lanes 0-15 rows, K {0..7,16..23}; lanes 16-31 K {8..15,24..31}
        const half_t* ap = actIn + row * ROWH + (kc + ks) * 32 + hi * 8;
        v8h s0 = *(const v8h*)ap;
        v8h s1 = *(const v8h*)(ap + 16);
        v16h a = __builtin_shufflevector(s0, s1, 0, 1, 2, 3, 4, 5, 6, 7,
                                         8, 9, 10, 11, 12, 13, 14, 15);
#pragma unroll
        for (int nt = 0; nt < NT; ++nt) {
          v16h b = *(const v16h*)(wbuf + (((size_t)ks * NT + nt) * 32 + lane) * 16);
          acc[nt] = wmma_f16(a, b, acc[nt]);
        }
      }
    }
  }

  const int col = lane & 15;
  if (MODE == 0) {
#pragma unroll
    for (int nt = 0; nt < NT; ++nt) {
      float bb = bias[nt * 16 + col];
#pragma unroll
      for (int r = 0; r < 8; ++r) {
        float v = acc[nt][r] + bb;
        v = v > 0.f ? v : 0.f;
        actOut[(hi * 8 + r) * ROWH + nt * 16 + col] = (half_t)v;
      }
    }
  } else if (MODE == 1) {
    float bb = bias[col];
    if (col == 0) {
#pragma unroll
      for (int r = 0; r < 8; ++r)
        gout[pointBase + hi * 8 + r] = acc[0][r] + bb;
    }
  } else {
    float bb = bias[col];
    if (col < 3) {
#pragma unroll
      for (int r = 0; r < 8; ++r) {
        float v = acc[0][r] + bb;
        gout[(size_t)(pointBase + hi * 8 + r) * 3 + col] = 1.f / (1.f + __expf(-v));
      }
    }
  }
}

// =======================================================================
// Full NeRF MLP: 128 threads = 4 waves, 64 points per block (16 per wave).
// Layer sequence fully unrolled with explicit LDS buffer arguments so the
// compiler can prove every activation pointer is LDS (ds_load, not flat).
// =======================================================================
__global__ __launch_bounds__(128, 1)
void nerf_mlp(const float* __restrict__ ro, const float* __restrict__ rd,
              const float* __restrict__ zv, int zPerPoint, int S,
              const half_t* __restrict__ wpack, const float* __restrict__ bias,
              float* __restrict__ densOut, float* __restrict__ rgbOut) {
  extern __shared__ char smemraw[];
  half_t* sm = (half_t*)smemraw;
  const int wave = threadIdx.x >> 5, lane = threadIdx.x & 31;
  half_t* wbuf = sm;
  half_t* bufA = sm + WBUF_H + wave * ACT_H;
  half_t* bufB = sm + WBUF_H + 4 * ACT_H + wave * ACT_H;
  half_t* peB  = sm + WBUF_H + 8 * ACT_H + wave * 2048;
  half_t* veB  = sm + WBUF_H + 8 * ACT_H + 4 * 2048 + wave * 1024;
  const int pointBase = blockIdx.x * 64 + wave * 16;
  const int row = lane & 15, hi = lane >> 4;

  // ---- positional / view encodings straight into LDS ----
  {
    int p = pointBase + row;
    int ray = p / S;
    int s = p - ray * S;
    float z = zPerPoint ? zv[p] : zv[s];
    float o0 = ro[ray * 3], o1 = ro[ray * 3 + 1], o2 = ro[ray * 3 + 2];
    float d0 = rd[ray * 3], d1 = rd[ray * 3 + 1], d2 = rd[ray * 3 + 2];
    float pt[3] = {o0 + d0 * z, o1 + d1 * z, o2 + d2 * z};
    float dv[3] = {d0, d1, d2};
    for (int c = hi * 64; c < hi * 64 + 64; ++c) {   // PD=123 padded to 128
      float v = 0.f;
      if (c < PDIM) {
        int g = c / 3, dim = c - 3 * g;
        if (g == 0) v = pt[dim];
        else if (g <= 20) v = __sinf(exp2f(10.f * (float)(g - 1) / 19.f) * pt[dim]);
        else              v = __cosf(exp2f(10.f * (float)(g - 21) / 19.f) * pt[dim]);
      }
      half_t h = (half_t)v;
      bufA[row * ROWH + c] = h;
      peB[row * 128 + c] = h;
    }
    for (int c = hi * 32; c < hi * 32 + 32; ++c) {   // VD=51 padded to 64
      float v = 0.f;
      if (c < VDIM) {
        int g = c / 3, dim = c - 3 * g;
        if (g == 0) v = dv[dim];
        else if (g <= 8) v = __sinf(exp2f(4.f * (float)(g - 1) / 7.f) * dv[dim]);
        else             v = __cosf(exp2f(4.f * (float)(g - 9) / 7.f) * dv[dim]);
      }
      veB[row * 64 + c] = (half_t)v;
    }
  }

  // layer 0 (K=128)
  gemm_tile<4, 16, 0>(wpack + WOFFg[0], bias + BOFFg[0], bufA, bufB, wbuf, nullptr, pointBase, lane);
  // layers 1..3 (ping-pong, explicit)
  gemm_tile<8, 16, 0>(wpack + WOFFg[1], bias + BOFFg[1], bufB, bufA, wbuf, nullptr, pointBase, lane);
  gemm_tile<8, 16, 0>(wpack + WOFFg[2], bias + BOFFg[2], bufA, bufB, wbuf, nullptr, pointBase, lane);
  gemm_tile<8, 16, 0>(wpack + WOFFg[3], bias + BOFFg[3], bufB, bufA, wbuf, nullptr, pointBase, lane);
  // skip concat: cols 256..383 <- pe  (x3 lives in bufA)
  for (int c = hi * 64; c < hi * 64 + 64; ++c)
    bufA[row * ROWH + 256 + c] = peB[row * 128 + c];
  // layer 4 (K=384)
  gemm_tile<12, 16, 0>(wpack + WOFFg[4], bias + BOFFg[4], bufA, bufB, wbuf, nullptr, pointBase, lane);
  // layers 5..7
  gemm_tile<8, 16, 0>(wpack + WOFFg[5], bias + BOFFg[5], bufB, bufA, wbuf, nullptr, pointBase, lane);
  gemm_tile<8, 16, 0>(wpack + WOFFg[6], bias + BOFFg[6], bufA, bufB, wbuf, nullptr, pointBase, lane);
  gemm_tile<8, 16, 0>(wpack + WOFFg[7], bias + BOFFg[7], bufB, bufA, wbuf, nullptr, pointBase, lane);
  // final x lives in bufA
  // density head (N padded to 16, col 0 valid)
  gemm_tile<8, 1, 1>(wpack + WOFFg[8], bias + BOFFg[8], bufA, nullptr, wbuf, densOut, pointBase, lane);
  // view concat: cols 256..319 <- ve
  for (int c = hi * 32; c < hi * 32 + 32; ++c)
    bufA[row * ROWH + 256 + c] = veB[row * 64 + c];
  // view0 (K=320, N=128), ReLU
  gemm_tile<10, 8, 0>(wpack + WOFFg[9], bias + BOFFg[9], bufA, bufB, wbuf, nullptr, pointBase, lane);
  // view1 -> sigmoid rgb (cols 0..2 valid)
  gemm_tile<4, 1, 2>(wpack + WOFFg[10], bias + BOFFg[10], bufB, nullptr, wbuf, rgbOut, pointBase, lane);
}

// =======================================================================
// Weight/bias packing into WMMA B-fragment layout (f16)
// =======================================================================
struct MlpPtrs {
  const float* posW[8]; const float* posB[8];
  const float* densW; const float* densB;
  const float* v0W; const float* v0B;
  const float* v1W; const float* v1B;
};

__global__ void pack_mlp(MlpPtrs mp, half_t* __restrict__ wdst, float* __restrict__ bdst) {
  int tid = blockIdx.x * blockDim.x + threadIdx.x;
  int nth = gridDim.x * blockDim.x;
  for (int idx = tid; idx < WTOTAL; idx += nth) {
    int g = 0;
    while (g < 10 && idx >= WOFFg[g + 1]) ++g;
    int e = idx - WOFFg[g];
    int NTn = NPg[g] >> 4;
    int t = e & 15, lane = (e >> 4) & 31, blk = e >> 9;
    int ntile = blk % NTn, kstep = blk / NTn;
    int k = kstep * 32 + (lane >> 4) * 8 + (t & 7) + (t >> 3) * 16;   // B: lanes 0-15 K {0..7,16..23}, lanes 16-31 K {8..15,24..31}
    int n = ntile * 16 + (lane & 15);
    const float* W = (g < 8) ? mp.posW[g] : (g == 8 ? mp.densW : (g == 9 ? mp.v0W : mp.v1W));
    float v = (k < DINg[g] && n < DOUTg[g]) ? W[(size_t)k * DOUTg[g] + n] : 0.f;
    wdst[idx] = (half_t)v;
  }
  for (int idx = tid; idx < BTOTAL; idx += nth) {
    int g = 0;
    while (g < 10 && idx >= BOFFg[g + 1]) ++g;
    int c = idx - BOFFg[g];
    const float* B = (g < 8) ? mp.posB[g] : (g == 8 ? mp.densB : (g == 9 ? mp.v0B : mp.v1B));
    bdst[idx] = (c < DOUTg[g]) ? B[c] : 0.f;
  }
}

// stratified z for coarse pass (shared across all rays, like the reference)
__global__ void zinit(float* __restrict__ zc) {
  int i = threadIdx.x;
  if (i >= NC) return;
  float lower = (i == 0) ? 0.f : ((float)i - 0.5f) / (float)(NC - 1);
  float upper = (i == NC - 1) ? 1.f : ((float)i + 0.5f) / (float)(NC - 1);
  zc[i] = lower + (upper - lower) * rnd01(42u, (unsigned)i);
}

// =======================================================================
// Volume integration (+ hierarchical fine-sample generation for coarse)
// =======================================================================
__global__ void integrate_kernel(const float* __restrict__ dens, const float* __restrict__ rgb,
                                 const float* __restrict__ zsrc, int zPerRay, int S,
                                 float* __restrict__ rgbOut, float* __restrict__ depthOut,
                                 float* __restrict__ wOut, float* __restrict__ zfineOut,
                                 int doSample) {
  int r = blockIdx.x * blockDim.x + threadIdx.x;
  if (r >= NUM_RAYS) return;
  const float* zb = zPerRay ? (zsrc + (size_t)r * S) : zsrc;
  float T = 1.f, c0 = 0.f, c1 = 0.f, c2 = 0.f, dep = 0.f;
  float wn[NC];
  for (int s = 0; s < S; ++s) {
    float z = zb[s];
    float dist = (s < S - 1) ? (zb[s + 1] - z) : 1e10f;
    float den = dens[(size_t)r * S + s];
    float a = 1.f - __expf(-den * dist);
    float w = a * T;
    T *= (1.f - a + 1e-10f);
    const float* rp = rgb + ((size_t)r * S + s) * 3;
    c0 += w * rp[0]; c1 += w * rp[1]; c2 += w * rp[2];
    dep += w * z;
    wOut[(size_t)r * S + s] = w;
    if (doSample && s < NC) wn[s] = w + 1e-5f;
  }
  rgbOut[r * 3] = c0; rgbOut[r * 3 + 1] = c1; rgbOut[r * 3 + 2] = c2;
  depthOut[r] = dep;
  if (!doSample) return;

  float total = 0.f;
  for (int s = 0; s < NC; ++s) total += wn[s];
  int cnt[NC - 1];
  for (int s = 0; s < NC - 1; ++s) cnt[s] = 0;
  for (int j = 0; j < NFI; ++j) {
    float u = rnd01((unsigned)r * 9781u + 0x9e3779b9u, (unsigned)j) * total;
    float acc = 0.f;
    int idx = NC - 2;
    for (int s = 0; s < NC; ++s) {
      acc += wn[s];
      if (acc > u) { idx = s; break; }
    }
    if (idx > NC - 2) idx = NC - 2;
    cnt[idx]++;
  }
  // merge sorted categorical samples (bin counts -> already sorted) with the
  // uniform linspace, keep every other of the 256 merged values.
  int bi = 0, bc = 0, ui = 0;
  for (int t = 0; t < 2 * NFI; ++t) {
    while (bi < NC - 1 && bc >= cnt[bi]) { bi++; bc = 0; }
    float za = (bi < NC - 1) ? 0.5f * (zb[bi] + zb[bi + 1]) : 3.0e38f;
    float zu = (ui < NFI) ? ((float)ui * (1.f / (float)(NFI - 1))) : 3.0e38f;
    float pick;
    if (za <= zu) { pick = za; bc++; } else { pick = zu; ui++; }
    if ((t & 1) == 0) zfineOut[(size_t)r * NFI + (t >> 1)] = pick;
  }
}

// =======================================================================
extern "C" void kernel_launch(void* const* d_in, const int* in_sizes, int n_in,
                              void* d_out, int out_size, void* d_ws, size_t ws_size,
                              hipStream_t stream) {
  (void)in_sizes; (void)n_in; (void)out_size;
  if (ws_size < WS_TOTAL) return;
  const float* rays_o = (const float*)d_in[0];
  const float* rays_d = (const float*)d_in[1];
  // jax tree order: coarse{density(W,b), pos[0..7](W,b), view0(W,b), view1(W,b)}, fine{...}
  auto mk = [&](int b) {
    MlpPtrs m;
    m.densW = (const float*)d_in[b];     m.densB = (const float*)d_in[b + 1];
    for (int i = 0; i < 8; ++i) {
      m.posW[i] = (const float*)d_in[b + 2 + 2 * i];
      m.posB[i] = (const float*)d_in[b + 3 + 2 * i];
    }
    m.v0W = (const float*)d_in[b + 18];  m.v0B = (const float*)d_in[b + 19];
    m.v1W = (const float*)d_in[b + 20];  m.v1B = (const float*)d_in[b + 21];
    return m;
  };
  MlpPtrs mc = mk(2), mf = mk(24);

  char* ws = (char*)d_ws;
  half_t* wpackC = (half_t*)(ws + OFF_WPACK);
  half_t* wpackF = wpackC + WTOTAL;
  float* biasC = (float*)(ws + OFF_BIAS);
  float* biasF = biasC + BTOTAL;
  float* zc = (float*)(ws + OFF_ZC);
  float* zf = (float*)(ws + OFF_ZF);
  float* densC = (float*)(ws + OFF_DC);
  float* rgbC  = (float*)(ws + OFF_RGBC);
  float* densF = (float*)(ws + OFF_DF);
  float* rgbF  = (float*)(ws + OFF_RGBF);
  float* out = (float*)d_out;

  zinit<<<1, 64, 0, stream>>>(zc);
  pack_mlp<<<1024, 256, 0, stream>>>(mc, wpackC, biasC);
  pack_mlp<<<1024, 256, 0, stream>>>(mf, wpackF, biasF);

  nerf_mlp<<<NUM_RAYS * NC / 64, 128, LDS_BYTES, stream>>>(
      rays_o, rays_d, zc, 0, NC, wpackC, biasC, densC, rgbC);
  integrate_kernel<<<NUM_RAYS / 128, 128, 0, stream>>>(
      densC, rgbC, zc, 0, NC, out + 0, out + 24576, out + 32768, zf, 1);
  nerf_mlp<<<NUM_RAYS * NFI / 64, 128, LDS_BYTES, stream>>>(
      rays_o, rays_d, zf, 1, NFI, wpackF, biasF, densF, rgbF);
  integrate_kernel<<<NUM_RAYS / 128, 128, 0, stream>>>(
      densF, rgbF, zf, 1, NFI, out + 12288, out + 28672, out + 294912, nullptr, 0);
}
